// BoxPredictor_65781719106273
// MI455X (gfx1250) — compile-verified, matrix operand order checked by reading, not run on hardware
//
#include <hip/hip_runtime.h>
#include <hip/hip_bf16.h>
#include <math.h>

typedef __attribute__((ext_vector_type(16))) _Float16 v16h;
typedef __attribute__((ext_vector_type(8)))  float    v8f;

#define BB 8
#define NN 4000
#define CC 91
#define BN (BB * NN)
#define REG_STRIDE (CC * 4)
#define DETS 100
#define NEGV (-1.0e9f)
#define CAP_MAX 32768

__device__ __forceinline__ float clampf(float x, float lo, float hi) {
    return fminf(fmaxf(x, lo), hi);
}

// ---------------------------------------------------------------------------
// Kernel 1: softmax (WMMA row-sum) + box decode + score/size filter + compact.
// One wave handles a 16-row tile. 128 threads = 4 waves/block.
// ---------------------------------------------------------------------------
__global__ void __launch_bounds__(128)
decode_score_filter(const float* __restrict__ logits,
                    const float* __restrict__ boxreg,
                    const float* __restrict__ props,
                    const int* __restrict__ hptr,
                    const int* __restrict__ wptr,
                    int* __restrict__ counts,
                    float* __restrict__ cboxes,
                    float* __restrict__ cscores,
                    float* __restrict__ clabels,
                    int cap)
{
    const float W = (float)(*wptr);
    const float H = (float)(*hptr);
    const float CLIP = 4.135166556742356f; // log(1000/16)

    const int tid  = threadIdx.x;
    const int lane = tid & 31;
    const int wid  = tid >> 5;
    const int tile = blockIdx.x * 4 + wid;
    const int m    = lane & 15;   // row within 16-row tile
    const int half = lane >> 4;   // which K-half this lane owns
    const int row  = tile * 16 + m;
    const bool rowok = (row < BN);

    const float* lg = logits + (size_t)row * CC;

    // Cache this lane's 48 logits (K-layout of 16-bit WMMA A-matrix):
    // element e of chunk holds K = e + 8*half + (e>=8 ? 8 : 0), col = chunk*32 + K
    float v[48];
    float mx = -3.0e38f;
#pragma unroll
    for (int chunk = 0; chunk < 3; ++chunk) {
#pragma unroll
        for (int e = 0; e < 16; ++e) {
            int K = e + 8 * half + (e >= 8 ? 8 : 0);
            int c = chunk * 32 + K;
            float val = (rowok && c < CC) ? lg[c] : -1.0e30f;
            v[chunk * 16 + e] = val;
            mx = fmaxf(mx, val);
        }
    }
    // full row max: combine the two half-lanes of this row
    mx = fmaxf(mx, __shfl_xor(mx, 16, 32));

#pragma unroll
    for (int i = 0; i < 48; ++i) {
        float t = v[i];
        v[i] = (t > -1.0e29f) ? __expf(t - mx) : 0.0f;
    }

    // WMMA row-sum reduction: D = A x Ones + C -> row sums replicated per column
    v16h ones;
#pragma unroll
    for (int e = 0; e < 16; ++e) ones[e] = (_Float16)1.0f;

    v8f acc = {};
#pragma unroll
    for (int chunk = 0; chunk < 3; ++chunk) {
        v16h a;
#pragma unroll
        for (int e = 0; e < 16; ++e) a[e] = (_Float16)v[chunk * 16 + e];
        acc = __builtin_amdgcn_wmma_f32_16x16x32_f16(
                  false, a, false, ones, (short)0, acc, false, false);
    }

    // Redistribute row sums: lane 0 holds rows 0..7 (VGPR j <-> M=j),
    // lane 16 holds rows 8..15.
    __shared__ float denom[4][16];
    if (lane == 0) {
#pragma unroll
        for (int j = 0; j < 8; ++j) denom[wid][j] = acc[j];
    }
    if (lane == 16) {
#pragma unroll
        for (int j = 0; j < 8; ++j) denom[wid][8 + j] = acc[j];
    }
    __syncthreads();
    if (!rowok) return;
    const float invden = 1.0f / denom[wid][m];

    const int b = row / NN;
    const float* pr = props + (size_t)row * 4;
    const float px1 = pr[0], py1 = pr[1], px2 = pr[2], py2 = pr[3];
    const float pw = px2 - px1, ph = py2 - py1;
    const float pcx = px1 + 0.5f * pw, pcy = py1 + 0.5f * ph;
    const float* rg = boxreg + (size_t)row * REG_STRIDE;

    float* myboxes  = cboxes  + (size_t)b * cap * 4;
    float* myscores = cscores + (size_t)b * cap;
    float* mylabels = clabels + (size_t)b * cap;

#pragma unroll
    for (int chunk = 0; chunk < 3; ++chunk) {
#pragma unroll
        for (int e = 0; e < 16; ++e) {
            int K = e + 8 * half + (e >= 8 ? 8 : 0);
            int c = chunk * 32 + K;
            if (c == 0 || c >= CC) continue;         // skip background + pad
            float score = v[chunk * 16 + e] * invden;
            if (score <= 0.05f) continue;
            float dx = rg[c * 4 + 0] * 0.1f;
            float dy = rg[c * 4 + 1] * 0.1f;
            float dw = fminf(rg[c * 4 + 2] * 0.2f, CLIP);
            float dh = fminf(rg[c * 4 + 3] * 0.2f, CLIP);
            float ncx = dx * pw + pcx;
            float ncy = dy * ph + pcy;
            float nw = __expf(dw) * pw;
            float nh = __expf(dh) * ph;
            float x1 = clampf(ncx - 0.5f * nw, 0.0f, W);
            float y1 = clampf(ncy - 0.5f * nh, 0.0f, H);
            float x2 = clampf(ncx + 0.5f * nw, 0.0f, W);
            float y2 = clampf(ncy + 0.5f * nh, 0.0f, H);
            if ((x2 - x1) >= 0.01f && (y2 - y1) >= 0.01f) {
                int idx = atomicAdd(&counts[b], 1);
                if (idx < cap) {
                    myboxes[idx * 4 + 0] = x1;
                    myboxes[idx * 4 + 1] = y1;
                    myboxes[idx * 4 + 2] = x2;
                    myboxes[idx * 4 + 3] = y2;
                    myscores[idx] = score;
                    mylabels[idx] = (float)c;
                }
            }
        }
    }
}

// ---------------------------------------------------------------------------
// Kernel 2: per-image greedy batched NMS (same-label IoU == class-offset trick).
// One 256-thread (8-wave) block per image.
// ---------------------------------------------------------------------------
__global__ void __launch_bounds__(256)
nms_kernel(const int* __restrict__ counts,
           const float* __restrict__ cboxes,
           float* __restrict__ cscores,
           const float* __restrict__ clabels,
           int cap,
           float* __restrict__ out)
{
    const int b   = blockIdx.x;
    const int bs  = blockDim.x;
    const int tid = threadIdx.x;
    const int lane = tid & 31;
    const int wid  = tid >> 5;

    int cnt = counts[b];
    if (cnt > cap) cnt = cap;

    const float* bx = cboxes + (size_t)b * cap * 4;
    float*       sc = cscores + (size_t)b * cap;
    const float* lb = clabels + (size_t)b * cap;

    float* out_boxes  = out;                    // [8][100][4]
    float* out_scores = out + BB * DETS * 4;    // [8][100]
    float* out_labels = out_scores + BB * DETS; // [8][100]
    float* out_keep   = out_labels + BB * DETS; // [8][100]

    __shared__ float s_val[8];
    __shared__ int   s_idx[8];
    __shared__ float c_box[4];
    __shared__ float c_lab, c_sc;
    __shared__ int   c_idx;

    for (int it = 0; it < DETS; ++it) {
        // --- block-wide argmax over surviving candidates ---
        float bv = NEGV;
        int bi = -1;
        for (int i = tid; i < cnt; i += bs) {
            float s = sc[i];
            if (s > bv) { bv = s; bi = i; }
        }
#pragma unroll
        for (int off = 16; off > 0; off >>= 1) {
            float ov = __shfl_down(bv, off, 32);
            int   oi = __shfl_down(bi, off, 32);
            if (ov > bv) { bv = ov; bi = oi; }
        }
        if (lane == 0) { s_val[wid] = bv; s_idx[wid] = bi; }
        __syncthreads();
        if (tid == 0) {
            float fv = s_val[0];
            int fi = s_idx[0];
            for (int w2 = 1; w2 < (bs >> 5); ++w2)
                if (s_val[w2] > fv) { fv = s_val[w2]; fi = s_idx[w2]; }
            if (fi >= 0 && fv > NEGV * 0.5f) {
                c_idx = fi; c_sc = fv;
                c_box[0] = bx[fi * 4 + 0]; c_box[1] = bx[fi * 4 + 1];
                c_box[2] = bx[fi * 4 + 2]; c_box[3] = bx[fi * 4 + 3];
                c_lab = lb[fi];
            } else {
                c_idx = -1;
            }
        }
        __syncthreads();

        const int ci = c_idx;
        if (ci < 0) {
            // exhausted: zero-fill remaining slots
            for (int s2 = it + tid; s2 < DETS; s2 += bs) {
                int o = b * DETS + s2;
                out_boxes[o * 4 + 0] = 0.0f; out_boxes[o * 4 + 1] = 0.0f;
                out_boxes[o * 4 + 2] = 0.0f; out_boxes[o * 4 + 3] = 0.0f;
                out_scores[o] = 0.0f; out_labels[o] = 0.0f; out_keep[o] = 0.0f;
            }
            break;
        }

        const float qx1 = c_box[0], qy1 = c_box[1], qx2 = c_box[2], qy2 = c_box[3];
        const float ql = c_lab;
        const float qa = (qx2 - qx1) * (qy2 - qy1);

        // --- suppress same-class boxes with IoU > 0.5 (includes self) ---
        for (int i = tid; i < cnt; i += bs) {
            if (lb[i] != ql) continue;
            float s = sc[i];
            if (s <= NEGV * 0.5f) continue;
            float x1 = bx[i * 4 + 0], y1 = bx[i * 4 + 1];
            float x2 = bx[i * 4 + 2], y2 = bx[i * 4 + 3];
            float ix1 = fmaxf(qx1, x1), iy1 = fmaxf(qy1, y1);
            float ix2 = fminf(qx2, x2), iy2 = fminf(qy2, y2);
            float inter = fmaxf(ix2 - ix1, 0.0f) * fmaxf(iy2 - iy1, 0.0f);
            float ar = (x2 - x1) * (y2 - y1);
            float iou = inter / fmaxf(qa + ar - inter, 1e-9f);
            if (iou > 0.5f) sc[i] = NEGV;
        }
        if (tid == 0) {
            sc[ci] = NEGV;
            int o = b * DETS + it;
            out_boxes[o * 4 + 0] = qx1; out_boxes[o * 4 + 1] = qy1;
            out_boxes[o * 4 + 2] = qx2; out_boxes[o * 4 + 3] = qy2;
            out_scores[o] = c_sc;
            out_labels[o] = ql;
            out_keep[o] = 1.0f;
        }
        __syncthreads(); // make suppression visible before next argmax
    }
}

// ---------------------------------------------------------------------------
extern "C" void kernel_launch(void* const* d_in, const int* in_sizes, int n_in,
                              void* d_out, int out_size, void* d_ws, size_t ws_size,
                              hipStream_t stream) {
    (void)in_sizes; (void)n_in; (void)out_size;

    const float* logits = (const float*)d_in[0];
    const float* boxreg = (const float*)d_in[1];
    const float* props  = (const float*)d_in[2];
    const int*   hptr   = (const int*)d_in[3];
    const int*   wptr   = (const int*)d_in[4];
    float* out = (float*)d_out;

    // workspace layout: [counts (256B header)] [boxes][scores][labels]
    const size_t header = 256;
    size_t cap = 0;
    if (ws_size > header)
        cap = (ws_size - header) / (size_t)(BB * 6 * sizeof(float)); // 24 B/cand x 8 imgs
    if (cap > CAP_MAX) cap = CAP_MAX;
    if (cap < 1) cap = 1;

    int*   counts  = (int*)d_ws;
    float* cboxes  = (float*)((char*)d_ws + header);
    float* cscores = cboxes + (size_t)BB * cap * 4;
    float* clabels = cscores + (size_t)BB * cap;

    hipMemsetAsync(counts, 0, BB * sizeof(int), stream);

    const int tiles = (BN + 15) / 16;          // 2000
    const int blocks1 = (tiles + 3) / 4;       // 4 waves/block
    decode_score_filter<<<blocks1, 128, 0, stream>>>(
        logits, boxreg, props, hptr, wptr,
        counts, cboxes, cscores, clabels, (int)cap);

    nms_kernel<<<BB, 256, 0, stream>>>(counts, cboxes, cscores, clabels, (int)cap, out);
}
